// featureMatcher_80384607911985
// MI455X (gfx1250) — matrix operand order repaired and checked by reading, not compiled, not run on hardware
//
#include <hip/hip_runtime.h>
#include <math.h>

typedef __attribute__((ext_vector_type(16))) _Float16 v16h;
typedef __attribute__((ext_vector_type(8)))  float    v8f;

#define B        16
#define N        1024
#define NP1      1025
#define DD       24
#define CSTR     1028                          // padded row stride (16B aligned)
#define UVSTR    1028                          // padded u/v stride (16B aligned)
#define CTOT     ((size_t)B * NP1 * CSTR)      // 16,859,200 floats (~67.4 MB)
#define UVTOT    (B * UVSTR)                   // 16448
#define SCALE    0.20412414523193154f          // 1/sqrt(24)
#define NORM     (-7.6246189861593985f)        // -ln(2048)
#define LOG_BIN  (-0.6931471805599453f)        // ln(1024) + NORM = -ln 2

// ---------------------------------------------------------------------------
// Kernel 1: scores GEMM via WMMA f16->f32, K=24 padded to 32.
// One wave computes a 16(n) x 64(m) strip: A loaded once, 4 B-tiles + 4 WMMAs.
// All loads are unconditional (clamped K) + register select, so they clause.
// ---------------------------------------------------------------------------
__global__ void __launch_bounds__(256)
gemm_scores(const float* __restrict__ dsc0, const float* __restrict__ dsc1,
            float* __restrict__ C) {
  const int lane = threadIdx.x & 31;
  const int hi   = lane >> 4;        // 0: lanes 0-15, 1: lanes 16-31
  const int ml   = lane & 15;

  const int w      = blockIdx.x * 8 + (threadIdx.x >> 5);  // global wave id
  const int mStrip = w & 15;          // 16 strips of 64 columns
  const int nTile  = (w >> 4) & 63;   // 64 row tiles
  const int b      = w >> 10;         // batch

  const int n0 = nTile * 16;
  const int m0 = mStrip * 64;
  const float* A0 = dsc0 + (size_t)b * DD * N;
  const float* B0 = dsc1 + (size_t)b * DD * N;

  // ---- A matrix (16x32 f16): lane ml is row M; element e -> K per ISA layout
  v16h a;
#pragma unroll
  for (int e = 0; e < 16; ++e) {
    const int K    = (e < 8) ? (hi * 8 + e) : (16 + hi * 8 + (e - 8));
    const bool ok  = (e < 8) || (hi == 0);     // K < 24
    const int  Kc  = ok ? K : 0;               // clamp: load always, select after
    float val = A0[(size_t)Kc * N + n0 + ml];
    val = ok ? val : 0.0f;
    a[e] = (_Float16)val;
  }

#pragma unroll
  for (int t = 0; t < 4; ++t) {
    const int jcol = m0 + t * 16 + ml;   // column index (also D-matrix column)
    // ---- B matrix (32x16 f16): lane ml is column N; element e -> K per ISA layout (same as A)
    v16h bb;
#pragma unroll
    for (int e = 0; e < 16; ++e) {
      const int  K  = (e < 8) ? (hi * 8 + e) : (16 + hi * 8 + (e - 8));
      const bool ok = (e < 8) || (hi == 0);    // K < 24
      const int  Kc = ok ? K : 0;
      float val = B0[(size_t)Kc * N + jcol];
      val = ok ? val : 0.0f;
      bb[e] = (_Float16)val;
    }

    v8f c = {};
    c = __builtin_amdgcn_wmma_f32_16x16x32_f16(
        /*neg_a=*/false, a, /*neg_b=*/false, bb,
        /*c_mod=*/(short)0, c, /*reuse_a=*/false, /*reuse_b=*/false);

    // D layout: VGPR r -> row M = r + 8*hi ; column = jcol
#pragma unroll
    for (int r = 0; r < 8; ++r) {
      int row = n0 + r + 8 * hi;
      C[((size_t)b * NP1 + row) * CSTR + jcol] = c[r] * SCALE;
    }
  }
}

// ---------------------------------------------------------------------------
// Kernel 2: fill dustbin row/column with alpha, zero-init u and v (incl. pads).
// ---------------------------------------------------------------------------
__global__ void fill_borders(float* __restrict__ C, float* __restrict__ u,
                             float* __restrict__ v,
                             const float* __restrict__ alpha_p) {
  int t = blockIdx.x * blockDim.x + threadIdx.x;
  if (t >= UVTOT) return;
  u[t] = 0.0f;
  v[t] = 0.0f;
  if (t < B * NP1) {
    float alpha = alpha_p[0];
    int b = t / NP1;
    int k = t % NP1;
    C[((size_t)b * NP1 + N) * CSTR + k] = alpha;   // last row
    C[((size_t)b * NP1 + k) * CSTR + N] = alpha;   // last column
  }
}

// ---------------------------------------------------------------------------
// Kernel 3: u[b,i] = log_mu[i] - logsumexp_j(C[b,i,j] + v[b,j])
// One wave per row; float4 loads (aligned via CSTR pad), per-vector max
// streaming logsumexp, shuffle reduce across the wave.
// ---------------------------------------------------------------------------
__global__ void __launch_bounds__(256)
row_update(const float* __restrict__ C, const float* __restrict__ v,
           float* __restrict__ u) {
  const int lane = threadIdx.x & 31;
  const int row  = blockIdx.x * 8 + (threadIdx.x >> 5);  // b*1025 + i
  const int b    = row / NP1;
  const int i    = row % NP1;
  const float*  Crow  = C + (size_t)row * CSTR;
  const float*  vb    = v + b * UVSTR;
  const float4* Crow4 = (const float4*)Crow;
  const float4* vb4   = (const float4*)vb;

  float m = -INFINITY, s = 0.0f;
#pragma unroll 4
  for (int t = lane; t < 256; t += 32) {       // 256 float4 = j 0..1023
    float4 cx = Crow4[t];
    float4 vx = vb4[t];
    float x0 = cx.x + vx.x, x1 = cx.y + vx.y;
    float x2 = cx.z + vx.z, x3 = cx.w + vx.w;
    float m4 = fmaxf(fmaxf(x0, x1), fmaxf(x2, x3));
    float nm = fmaxf(m, m4);
    s = s * __expf(m - nm) + __expf(x0 - nm) + __expf(x1 - nm)
                           + __expf(x2 - nm) + __expf(x3 - nm);
    m = nm;
  }
  if (lane == 0) {                              // tail element j = 1024
    float x  = Crow[N] + vb[N];
    float nm = fmaxf(m, x);
    s = s * __expf(m - nm) + __expf(x - nm);
    m = nm;
  }
#pragma unroll
  for (int off = 16; off > 0; off >>= 1) {
    float om = __shfl_xor(m, off);
    float os = __shfl_xor(s, off);
    float nm = fmaxf(m, om);
    s = s * __expf(m - nm) + os * __expf(om - nm);
    m = nm;
  }
  if (lane == 0) {
    float log_mu = (i == N) ? LOG_BIN : NORM;
    u[b * UVSTR + i] = log_mu - (m + __logf(s));
  }
}

// ---------------------------------------------------------------------------
// Kernel 4: v[b,j] = log_nu[j] - logsumexp_i(C[b,i,j] + u[b,i])
// Workgroup (8 waves) owns 32 consecutive columns; each wave streams a
// 128/129-row chunk, 4 rows at a time (clause'd coalesced loads, one max4 +
// one rescale per 4 elements); partials merged through LDS.
// ---------------------------------------------------------------------------
__global__ void __launch_bounds__(256)
col_update(const float* __restrict__ C, const float* __restrict__ u,
           float* __restrict__ v) {
  __shared__ float sM[8][32];
  __shared__ float sS[8][32];

  const int lane = threadIdx.x & 31;
  const int wave = threadIdx.x >> 5;
  const int b    = blockIdx.x / 33;
  const int g    = blockIdx.x % 33;
  const int j    = g * 32 + lane;
  const int jc   = (j < NP1) ? j : (NP1 - 1);   // clamp OOB lanes (discarded)

  const float* Cb = C + (size_t)b * NP1 * CSTR;
  const float* ub = u + b * UVSTR;

  const int i0   = wave * 128;
  const int iend = (wave == 7) ? NP1 : (i0 + 128);

  float m = -INFINITY, s = 0.0f;
  int i = i0;
  for (; i + 3 < iend; i += 4) {
    float x0 = Cb[(size_t)(i + 0) * CSTR + jc] + ub[i + 0];
    float x1 = Cb[(size_t)(i + 1) * CSTR + jc] + ub[i + 1];
    float x2 = Cb[(size_t)(i + 2) * CSTR + jc] + ub[i + 2];
    float x3 = Cb[(size_t)(i + 3) * CSTR + jc] + ub[i + 3];
    float m4 = fmaxf(fmaxf(x0, x1), fmaxf(x2, x3));
    float nm = fmaxf(m, m4);
    s = s * __expf(m - nm) + __expf(x0 - nm) + __expf(x1 - nm)
                           + __expf(x2 - nm) + __expf(x3 - nm);
    m = nm;
  }
  for (; i < iend; ++i) {
    float x  = Cb[(size_t)i * CSTR + jc] + ub[i];
    float nm = fmaxf(m, x);
    s = s * __expf(m - nm) + __expf(x - nm);
    m = nm;
  }
  sM[wave][lane] = m;
  sS[wave][lane] = s;
  __syncthreads();

  if (threadIdx.x < 32) {
    float M = -INFINITY, S = 0.0f;
#pragma unroll
    for (int w2 = 0; w2 < 8; ++w2) {
      float om = sM[w2][lane];
      float os = sS[w2][lane];
      float nm = fmaxf(M, om);
      S = S * __expf(M - nm) + os * __expf(om - nm);
      M = nm;
    }
    if (j < NP1) {
      float log_nu = (j == N) ? LOG_BIN : NORM;
      v[b * UVSTR + j] = log_nu - (M + __logf(S));
    }
  }
}

// ---------------------------------------------------------------------------
// Kernel 5: Z = C + u[:,i] + v[:,j] - norm  -> d_out (unpadded [16,1025,1025])
// ---------------------------------------------------------------------------
__global__ void __launch_bounds__(256)
finalize_z(const float* __restrict__ C, const float* __restrict__ u,
           const float* __restrict__ v, float* __restrict__ out) {
  const int row = blockIdx.y;                 // b*1025 + i
  const int j   = blockIdx.x * 256 + threadIdx.x;
  if (j >= NP1) return;
  const int b = row / NP1;
  const int i = row % NP1;
  out[(size_t)row * NP1 + j] =
      C[(size_t)row * CSTR + j] + u[b * UVSTR + i] + v[b * UVSTR + j] - NORM;
}

// ---------------------------------------------------------------------------
extern "C" void kernel_launch(void* const* d_in, const int* in_sizes, int n_in,
                              void* d_out, int out_size, void* d_ws, size_t ws_size,
                              hipStream_t stream) {
  const float* dsc0  = (const float*)d_in[0];
  const float* dsc1  = (const float*)d_in[1];
  const float* alpha = (const float*)d_in[2];

  float* C = (float*)d_ws;           // padded couplings (~67.4 MB, L2-resident)
  float* u = C + CTOT;               // 16448 floats
  float* v = u + UVTOT;              // 16448 floats
  float* out = (float*)d_out;

  // 1) scores -> interior of couplings (WMMA f16/f32)
  gemm_scores<<<2048, 256, 0, stream>>>(dsc0, dsc1, C);
  // 2) dustbin borders + u/v init
  fill_borders<<<(UVTOT + 255) / 256, 256, 0, stream>>>(C, u, v, alpha);
  // 3) 20 Sinkhorn iterations (L2-resident sweeps)
  for (int it = 0; it < 20; ++it) {
    row_update<<<2050, 256, 0, stream>>>(C, v, u);
    col_update<<<16 * 33, 256, 0, stream>>>(C, u, v);
  }
  // 4) final transport plan
  finalize_z<<<dim3((NP1 + 255) / 256, B * NP1), 256, 0, stream>>>(C, u, v, out);
}
